// SwarmNet_53077205844510
// MI455X (gfx1250) — matrix operand order repaired and checked by reading, not compiled, not run on hardware
//
#include <hip/hip_runtime.h>

// ---------------- WMMA helpers (gfx1250, wave32) ----------------
typedef __attribute__((ext_vector_type(16))) _Float16 v16h;
typedef __attribute__((ext_vector_type(8)))  _Float16 v8h;
typedef __attribute__((ext_vector_type(8)))  float    v8f;

__device__ __forceinline__ v8f wmma_f16(v16h a, v16h b, v8f c) {
    return __builtin_amdgcn_wmma_f32_16x16x32_f16(false, a, false, b, (short)0, c, false, false);
}

// A fragment: 16x32 f16 tile from row-major matrix (row stride `ld` halves).
__device__ __forceinline__ v16h load_a_frag(const _Float16* base, int ld, int k, int lane) {
    int m = lane & 15, hi = lane >> 4;
    const _Float16* p = base + (size_t)m * ld + k + hi * 8;
    v8h lo = *(const v8h*)p;
    v8h hv = *(const v8h*)(p + 16);
    return __builtin_shufflevector(lo, hv, 0,1,2,3,4,5,6,7,8,9,10,11,12,13,14,15);
}

// B fragment: 32x16 f16 tile; weights stored transposed WT[N][K] so each lane
// reads halves k+8*hi..+7 and k+16+8*hi..+7 of its column (same k order as A).
__device__ __forceinline__ v16h load_b_frag(const _Float16* baseT, int ld, int k, int lane) {
    int n = lane & 15, hi = lane >> 4;
    const _Float16* p = baseT + (size_t)n * ld + k + hi * 8;
    v8h lo = *(const v8h*)p;
    v8h hv = *(const v8h*)(p + 16);
    return __builtin_shufflevector(lo, hv, 0,1,2,3,4,5,6,7,8,9,10,11,12,13,14,15);
}

// ---------------- model constants ----------------
#define NB     64
#define NA     64
#define ND     4
#define NSTEP  8
#define TSEG   7
#define TMAX   (TSEG + NSTEP)     // 15
#define NNODE  (NB * NA)          // 4096
#define NSK    288                // 260 padded to 9*32

// ---------------- weight prep ----------------
__global__ void prep_ew0(const float* __restrict__ ew0, _Float16* __restrict__ ew0catT) {
    int idx = blockIdx.x * 256 + threadIdx.x;
    if (idx >= 512 * 128) return;
    int n = idx >> 7, k = idx & 127;
    float v = (n < 256) ? ew0[(size_t)k * 256 + n] : ew0[(size_t)(128 + k) * 256 + (n - 256)];
    ew0catT[idx] = (_Float16)v;
}

// WT[n][kp] = (kp<K) ? W[kp][n] : 0   (W is [K][N] row-major; WT is [N][Kpad])
__global__ void prep_T(const float* __restrict__ W, _Float16* __restrict__ WT,
                       int K, int N, int Kpad) {
    int idx = blockIdx.x * 256 + threadIdx.x;
    if (idx >= N * Kpad) return;
    int n = idx / Kpad, k = idx % Kpad;
    WT[idx] = (k < K) ? (_Float16)W[(size_t)k * N + n] : (_Float16)0.f;
}

// time_segs [B][7][N][4] -> segs [B][N][15][4] rows 0..6
__global__ void seg_init(const float* __restrict__ ts, float* __restrict__ segs) {
    int idx = blockIdx.x * 256 + threadIdx.x;
    if (idx >= NB * TSEG * NA * ND) return;
    int d = idx & 3;
    int n = (idx >> 2) & 63;
    int bt = idx >> 8;           // b*7 + t
    int t = bt % TSEG, b = bt / TSEG;
    segs[(((size_t)b * NA + n) * TMAX + t) * ND + d] = ts[idx];
}

// ---------------- conv encoder (fp32 VALU; ~0.5 GMAC/step) ----------------
__global__ void encoder_kernel(const float* __restrict__ segs,
                               const float* __restrict__ cw0, const float* __restrict__ cb0,
                               const float* __restrict__ cw1, const float* __restrict__ cb1,
                               const float* __restrict__ cw2, const float* __restrict__ cb2,
                               _Float16* __restrict__ H, int step) {
    __shared__ float x[TSEG][ND];
    __shared__ float c0s[5][64];
    __shared__ float c1s[3][128];
    int node = blockIdx.x;
    int tid = threadIdx.x;        // 128 threads
    const float* s = segs + ((size_t)node * TMAX + step) * ND;
    if (tid < TSEG * ND) x[tid >> 2][tid & 3] = s[tid];
    __syncthreads();
    {   // conv0: 5 pos x 64 filters, K=3x4
        int f = tid & 63;
        for (int p = tid >> 6; p < 5; p += 2) {
            float acc = cb0[f];
#pragma unroll
            for (int k = 0; k < 3; ++k)
#pragma unroll
                for (int c = 0; c < 4; ++c)
                    acc += x[p + k][c] * cw0[(k * 4 + c) * 64 + f];
            c0s[p][f] = fmaxf(acc, 0.f);
        }
    }
    __syncthreads();
    for (int p = 0; p < 3; ++p) {  // conv1
        float acc = cb1[tid];
        for (int k = 0; k < 3; ++k)
            for (int c = 0; c < 64; ++c)
                acc += c0s[p + k][c] * cw1[(k * 64 + c) * 128 + tid];
        c1s[p][tid] = fmaxf(acc, 0.f);
    }
    __syncthreads();
    {   // conv2
        float acc = cb2[tid];
        for (int k = 0; k < 3; ++k)
            for (int c = 0; c < 128; ++c)
                acc += c1s[k][c] * cw2[(k * 128 + c) * 128 + tid];
        H[(size_t)node * 128 + tid] = (_Float16)fmaxf(acc, 0.f);
    }
}

// ---------------- generic WMMA GEMM: out = [relu](A @ BT^T + bias) ----------------
// 128 threads = 4 waves; wave tile = 32 rows x 64 cols (2 A-frags reuse each B-frag).
// Block tile 128x64, grid = (M/128, N/64).
__global__ void gemm_kernel(const _Float16* __restrict__ A, const _Float16* __restrict__ BT,
                            const float* __restrict__ bias, void* __restrict__ out,
                            int M, int N, int K, int relu, int out_f16) {
    int w = threadIdx.x >> 5, lane = threadIdx.x & 31;
    int m0 = blockIdx.x * 128 + w * 32;
    int n0 = blockIdx.y * 64;
    v8f acc[2][4] = {};
    const _Float16* arow0 = A + (size_t)m0 * K;
    const _Float16* arow1 = arow0 + (size_t)16 * K;
    for (int k = 0; k < K; k += 32) {
        v16h a0 = load_a_frag(arow0, K, k, lane);
        v16h a1 = load_a_frag(arow1, K, k, lane);
#pragma unroll
        for (int j = 0; j < 4; ++j) {
            v16h b = load_b_frag(BT + (size_t)(n0 + j * 16) * K, K, k, lane);
            acc[0][j] = wmma_f16(a0, b, acc[0][j]);
            acc[1][j] = wmma_f16(a1, b, acc[1][j]);
        }
    }
    int hi = lane >> 4, nn = lane & 15;
#pragma unroll
    for (int i = 0; i < 2; ++i)
#pragma unroll
        for (int j = 0; j < 4; ++j) {
            int col = n0 + j * 16 + nn;
            float bv = bias ? bias[col] : 0.f;
#pragma unroll
            for (int v = 0; v < 8; ++v) {
                int row = m0 + i * 16 + v + 8 * hi;
                float val = acc[i][j][v] + bv;
                if (relu) val = fmaxf(val, 0.f);
                if (out_f16) ((_Float16*)out)[(size_t)row * N + col] = (_Float16)val;
                else         ((float*)out)[(size_t)row * N + col] = val;
            }
        }
}

// ---------------- fused edge MLP layer 2 + scatter-add (dense graph) ----------------
// Per block: batch b, 16 targets. Sources in chunks of 8 (128 edges/chunk):
//   e1 = relu(hs[s] + ht[t] + eb0)  -> LDS f16 tile (128x256)
//   e2 = relu(e1 @ ew1 + eb1), diagonal (s==t) masked, accumulated per target in regs.
// Wave tile: 2 source M-tiles x 8 N-tiles (B-frag feeds 2 WMMAs -> 16B/lane/WMMA global).
// Deterministic: the 4 waves sharing a column half reduce through disjoint LDS slabs.
__global__ void edge_kernel(const float* __restrict__ HSHT, const float* __restrict__ eb0,
                            const _Float16* __restrict__ ewt1, const float* __restrict__ eb1,
                            const float* __restrict__ segs, _Float16* __restrict__ NS, int step) {
    __shared__ float    sht[16][264];                // ht[t] + eb0
    __shared__ _Float16 sE1[128][272];               // e1 tile (padded vs bank conflicts)
    __shared__ float    snm4[4][16][264];            // per-wave-group partials
    int b  = blockIdx.x;
    int t0 = blockIdx.y * 16;
    int tid = threadIdx.x;                           // 256 threads = 8 waves

    for (int idx = tid; idx < 16 * 256; idx += 256) {
        int r = idx >> 8, c = idx & 255;
        sht[r][c] = HSHT[((size_t)(b * NA + t0 + r)) * 512 + 256 + c] + eb0[c];
    }
    __syncthreads();

    int w = tid >> 5, lane = tid & 31;
    int spair = (w >> 1) * 2;                        // first of this wave's 2 sources in chunk
    int nbase = (w & 1) * 128;                       // column half handled by this wave
    int nn = lane & 15, hi = lane >> 4;
    float ebv[8];
#pragma unroll
    for (int j = 0; j < 8; ++j) ebv[j] = eb1[nbase + j * 16 + nn];

    v8f nmacc[8] = {};
    int e  = tid >> 1;                               // edge row 0..127 (= s_local*16 + t_local)
    int cb = (tid & 1) * 128;
    int sl = e >> 4, tl = e & 15;

    for (int s0 = 0; s0 < NA; s0 += 8) {
        const float* hs = HSHT + ((size_t)(b * NA + s0 + sl)) * 512;   // hs half
#pragma unroll 4
        for (int c = 0; c < 128; ++c) {
            float v = hs[cb + c] + sht[tl][cb + c];
            sE1[e][cb + c] = (_Float16)fmaxf(v, 0.f);
        }
        __syncthreads();

        v8f acc[2][8] = {};
        const _Float16* abase0 = &sE1[(spair + 0) * 16][0];
        const _Float16* abase1 = &sE1[(spair + 1) * 16][0];
        for (int k = 0; k < 256; k += 32) {
            v16h a0 = load_a_frag(abase0, 272, k, lane);
            v16h a1 = load_a_frag(abase1, 272, k, lane);
#pragma unroll
            for (int j = 0; j < 8; ++j) {
                v16h bf = load_b_frag(ewt1 + (size_t)(nbase + j * 16) * 256, 256, k, lane);
                acc[0][j] = wmma_f16(a0, bf, acc[0][j]);
                acc[1][j] = wmma_f16(a1, bf, acc[1][j]);
            }
        }
#pragma unroll
        for (int i = 0; i < 2; ++i) {
            int s_glob = s0 + spair + i;
#pragma unroll
            for (int j = 0; j < 8; ++j)
#pragma unroll
                for (int v = 0; v < 8; ++v) {
                    int r = v + 8 * hi;              // target row within tile
                    float val = fmaxf(acc[i][j][v] + ebv[j], 0.f);
                    if (s_glob == t0 + r) val = 0.f; // mask diagonal edge
                    nmacc[j][v] += val;
                }
        }
        __syncthreads();                             // protect sE1 before next chunk
    }

#pragma unroll
    for (int j = 0; j < 8; ++j)
#pragma unroll
        for (int v = 0; v < 8; ++v)
            snm4[w >> 1][v + 8 * hi][nbase + j * 16 + nn] = nmacc[j][v];
    __syncthreads();

    // NS = [prev | nm | 0-pad], nm = sum of 4 slabs (deterministic order)
    for (int idx = tid; idx < 16 * NSK; idx += 256) {
        int r = idx / NSK, c = idx % NSK;
        size_t node = (size_t)b * NA + t0 + r;
        float val;
        if (c < 4)        val = segs[(node * TMAX + step + 6) * ND + c];
        else if (c < 260) {
            int cc = c - 4;
            val = snm4[0][r][cc] + snm4[1][r][cc] + snm4[2][r][cc] + snm4[3][r][cc];
        } else            val = 0.f;
        NS[node * NSK + c] = (_Float16)val;
    }
}

// ---------------- output head: out = D1 @ ow + ob + prev (N=4, fp32 VALU) ----------------
__global__ void out_kernel(const _Float16* __restrict__ D1, const float* __restrict__ ow,
                           const float* __restrict__ ob, float* __restrict__ segs,
                           float* __restrict__ dout, int step) {
    int idx = blockIdx.x * 256 + threadIdx.x;        // 4096 nodes * 4 dims
    if (idx >= NNODE * ND) return;
    int node = idx >> 2, d = idx & 3;
    const _Float16* row = D1 + (size_t)node * 256;
    float acc = ob[d];
    for (int k = 0; k < 256; ++k) acc += (float)row[k] * ow[k * 4 + d];
    float prev = segs[((size_t)node * TMAX + step + 6) * ND + d];
    float val = acc + prev;
    segs[((size_t)node * TMAX + step + 7) * ND + d] = val;
    int b = node >> 6, n = node & 63;
    dout[(((size_t)b * NSTEP + step) * NA + n) * ND + d] = val;
}

// ---------------- host ----------------
extern "C" void kernel_launch(void* const* d_in, const int* in_sizes, int n_in,
                              void* d_out, int out_size, void* d_ws, size_t ws_size,
                              hipStream_t stream) {
    const float* time_segs = (const float*)d_in[0];
    const float* cw0 = (const float*)d_in[1];
    const float* cb0 = (const float*)d_in[2];
    const float* cw1 = (const float*)d_in[3];
    const float* cb1 = (const float*)d_in[4];
    const float* cw2 = (const float*)d_in[5];
    const float* cb2 = (const float*)d_in[6];
    const float* ew0 = (const float*)d_in[7];
    const float* eb0 = (const float*)d_in[8];
    const float* ew1 = (const float*)d_in[9];
    const float* eb1 = (const float*)d_in[10];
    const float* dw0 = (const float*)d_in[11];
    const float* db0 = (const float*)d_in[12];
    const float* dw1 = (const float*)d_in[13];
    const float* db1 = (const float*)d_in[14];
    const float* ow  = (const float*)d_in[15];
    const float* ob  = (const float*)d_in[16];
    // d_in[17]/[18] = src/tgt: unused — graph is dense (ones - eye), handled analytically.

    char* ws = (char*)d_ws;
    size_t off = 0;
    auto alloc = [&](size_t bytes) {
        void* p = ws + off;
        off = (off + bytes + 255) & ~(size_t)255;
        return p;
    };
    float*    segs    = (float*)   alloc((size_t)NNODE * TMAX * ND * 4);
    _Float16* H       = (_Float16*)alloc((size_t)NNODE * 128 * 2);
    float*    HSHT    = (float*)   alloc((size_t)NNODE * 512 * 4);
    _Float16* NS      = (_Float16*)alloc((size_t)NNODE * NSK * 2);
    _Float16* D0      = (_Float16*)alloc((size_t)NNODE * 256 * 2);
    _Float16* D1      = (_Float16*)alloc((size_t)NNODE * 256 * 2);
    _Float16* ew0catT = (_Float16*)alloc((size_t)512 * 128 * 2);
    _Float16* ewt1    = (_Float16*)alloc((size_t)256 * 256 * 2);
    _Float16* dw0pT   = (_Float16*)alloc((size_t)256 * NSK * 2);
    _Float16* dw1T    = (_Float16*)alloc((size_t)256 * 256 * 2);
    (void)ws_size; (void)in_sizes; (void)n_in; (void)out_size;

    seg_init<<<(NB * TSEG * NA * ND + 255) / 256, 256, 0, stream>>>(time_segs, segs);
    prep_ew0<<<(512 * 128 + 255) / 256, 256, 0, stream>>>(ew0, ew0catT);
    prep_T<<<(256 * 256 + 255) / 256, 256, 0, stream>>>(ew1, ewt1, 256, 256, 256);
    prep_T<<<(256 * NSK + 255) / 256, 256, 0, stream>>>(dw0, dw0pT, 260, 256, NSK);
    prep_T<<<(256 * 256 + 255) / 256, 256, 0, stream>>>(dw1, dw1T, 256, 256, 256);

    for (int step = 0; step < NSTEP; ++step) {
        encoder_kernel<<<NNODE, 128, 0, stream>>>(segs, cw0, cb0, cw1, cb1, cw2, cb2, H, step);
        // [hs | ht] = H @ [ew0a | ew0b] : M=4096, N=512, K=128, f32 out, no bias/relu
        gemm_kernel<<<dim3(NNODE / 128, 512 / 64), 128, 0, stream>>>(
            H, ew0catT, nullptr, HSHT, NNODE, 512, 128, 0, 0);
        edge_kernel<<<dim3(NB, NA / 16), 256, 0, stream>>>(HSHT, eb0, ewt1, eb1, segs, NS, step);
        // decoder layer 0: M=4096, N=256, K=288(padded 260), relu, f16 out
        gemm_kernel<<<dim3(NNODE / 128, 256 / 64), 128, 0, stream>>>(
            NS, dw0pT, db0, D0, NNODE, 256, NSK, 1, 1);
        // decoder layer 1: K=256, relu, f16 out
        gemm_kernel<<<dim3(NNODE / 128, 256 / 64), 128, 0, stream>>>(
            D0, dw1T, db1, D1, NNODE, 256, 256, 1, 1);
        out_kernel<<<(NNODE * ND + 255) / 256, 256, 0, stream>>>(
            D1, ow, ob, segs, (float*)d_out, step);
    }
}
